// SGTA_61203283968637
// MI455X (gfx1250) — compile-verified, matrix operand order checked by reading, not run on hardware
//
#include <hip/hip_runtime.h>
#include <hip/hip_bf16.h>

// ---------------------------------------------------------------------------
// MI455X (gfx1250) channel-attention (XCA-style) block.
//
// x[8,192,128,128] f32; w_qkv[576,192]; w_dw[576,1,3,3]; temperature[8];
// w_proj[192,192] -> out [8,192,128,128] f32.
//
// ~43 GFLOP vs ~850 MB moved with f16 intermediates -> memory bound at
// 23.3 TB/s (~37us). All GEMMs via v_wmma_f32_16x16x32_f16 (f32 accum).
// Fusion: out = (W_proj @ blockdiag(attn)) @ V  -> one final GEMM.
// L2-normalization folded into softmax as reciprocal-row-norm scaling.
// ---------------------------------------------------------------------------

typedef __attribute__((ext_vector_type(16))) _Float16 v16h;
typedef __attribute__((ext_vector_type(8)))  _Float16 v8h;
typedef __attribute__((ext_vector_type(8)))  float    v8f;
typedef __attribute__((ext_vector_type(4)))  float    v4f;

#define DIMC   192
#define HEADS  8
#define HD     24          // head dim
#define HW     16384       // 128*128
#define BATCH  8
#define C3     576         // 3*DIMC

// ---------------------------------------------------------------------------
// GEMM: D[b][M,N] = A[b][M,K=192] * B[b][K=192,N]   (row-major everywhere)
// A f32 (converted to f16 per lane). B f32 or f16 (BH). D f16 or f32 (OH).
// One wave per block -> 64(M) x 16(N) tile; B tile staged via LDS so global
// B reads are coalesced and the LDS gather builds the WMMA B-fragment
// (column-per-lane) layout.
//
// WMMA 16-bit A 16x32 layout (ISA 7.12.2): lane<16 -> M=lane, halves[0..7] =
// K 0..7, halves[8..15] = K 16..23 ; lane>=16 -> M=lane-16, K 8..15 / 24..31.
// B 32x16 mirrors this with N in place of M. f32 C/D: lane<16 -> N=lane,
// vgpr r -> M=r ; lane>=16 -> M=8+r.
// ---------------------------------------------------------------------------
template <bool BH, bool OH>
__global__ __launch_bounds__(32) void gemm192_wmma(
    const float* __restrict__ A, long aStride,
    const void*  __restrict__ Bv, long bStride,
    void*        __restrict__ Dv, long dStride,
    int N)
{
  constexpr int K = DIMC;
  __shared__ _Float16 bt[32 * 16];          // B tile: 32 (K) x 16 (N), f16

  const int lane = threadIdx.x;             // wave32
  const int ln   = lane & 15;
  const int kb   = (lane < 16) ? 0 : 8;     // per-lane K sub-offset
  const long n0  = (long)blockIdx.x * 16;
  const int  m0  = blockIdx.y * 64;
  const int  bb  = blockIdx.z;

  const float* Ab = A + (long)bb * aStride;
  v8f acc[4] = {};

  for (int k0 = 0; k0 < K; k0 += 32) {
    // ---- stage B tile: lane t loads row k0+t, 16 contiguous elements ----
    if constexpr (BH) {
      const _Float16* Bb = (const _Float16*)Bv + (long)bb * bStride
                           + (long)(k0 + lane) * N + n0;
      if (k0 + 32 < K) __builtin_prefetch(Bb + 32L * N, 0, 1); // global_prefetch_b8
      const v8h* bp = (const v8h*)Bb;
      *((v8h*)&bt[lane * 16])     = bp[0];
      *((v8h*)&bt[lane * 16 + 8]) = bp[1];
    } else {
      const float* Bb = (const float*)Bv + (long)bb * bStride
                        + (long)(k0 + lane) * N + n0;
      if (k0 + 32 < K) __builtin_prefetch(Bb + 32L * N, 0, 1); // global_prefetch_b8
      const v4f* bp = (const v4f*)Bb;
      v4f f0 = bp[0], f1 = bp[1], f2 = bp[2], f3 = bp[3];
      v8h h0, h1;
#pragma unroll
      for (int i = 0; i < 4; ++i) {
        h0[i]     = (_Float16)f0[i];  h0[4 + i] = (_Float16)f1[i];
        h1[i]     = (_Float16)f2[i];  h1[4 + i] = (_Float16)f3[i];
      }
      *((v8h*)&bt[lane * 16])     = h0;
      *((v8h*)&bt[lane * 16 + 8]) = h1;
    }
    __syncthreads();   // cross-lane LDS visibility (single wave: cheap)

    // ---- gather WMMA B fragment (column ln, K rows kb..kb+7 / +16) ----
    v16h bf;
#pragma unroll
    for (int i = 0; i < 8; ++i) {
      bf[i]     = bt[(kb + i) * 16 + ln];
      bf[8 + i] = bt[(kb + 16 + i) * 16 + ln];
    }

    // ---- 4 M-subtiles reuse the B fragment ----
#pragma unroll
    for (int s = 0; s < 4; ++s) {
      const int r = m0 + s * 16 + ln;                 // A row (M multiple of 64)
      const v4f* ap = (const v4f*)(Ab + (long)r * K + k0 + kb);
      v4f a0 = ap[0], a1 = ap[1], a2 = ap[4], a3 = ap[5];   // K: kb..+7, kb+16..+23
      v16h af;
#pragma unroll
      for (int i = 0; i < 4; ++i) {
        af[i]      = (_Float16)a0[i];  af[4 + i]  = (_Float16)a1[i];
        af[8 + i]  = (_Float16)a2[i];  af[12 + i] = (_Float16)a3[i];
      }
      acc[s] = __builtin_amdgcn_wmma_f32_16x16x32_f16(
          false, af, false, bf, (short)0, acc[s], false, false);
    }
    // no trailing barrier needed: single wave, DS ops in-order per wave
  }

  // ---- store D ----
  const long col = n0 + ln;
#pragma unroll
  for (int s = 0; s < 4; ++s) {
    const int rbase = m0 + s * 16 + ((lane < 16) ? 0 : 8);
    if constexpr (OH) {
      _Float16* Db = (_Float16*)Dv + (long)blockIdx.z * dStride;
#pragma unroll
      for (int r = 0; r < 8; ++r)
        Db[(long)(rbase + r) * N + col] = (_Float16)acc[s][r];
    } else {
      float* Db = (float*)Dv + (long)blockIdx.z * dStride;
#pragma unroll
      for (int r = 0; r < 8; ++r)
        Db[(long)(rbase + r) * N + col] = acc[s][r];
    }
  }
}

// ---------------------------------------------------------------------------
// Zero the sumsq accumulator (ws is not re-poisoned between replays).
// ---------------------------------------------------------------------------
__global__ __launch_bounds__(256) void zero_kernel(float* __restrict__ p, int n)
{
  const int i = blockIdx.x * 256 + threadIdx.x;
  if (i < n) p[i] = 0.f;
}

// ---------------------------------------------------------------------------
// Depthwise 3x3 conv (SAME), f16 in/out, f32 math. Each 256-thread block
// covers 256 consecutive pixels of ONE channel, so it also block-reduces the
// sum of squares and atomically accumulates the q/k row norms (fusing the
// normalization pre-pass and saving a full 100MB read).
// ---------------------------------------------------------------------------
__global__ __launch_bounds__(256) void dwconv3x3_kernel(
    const _Float16* __restrict__ in, const float* __restrict__ w,
    _Float16* __restrict__ out, float* __restrict__ sumsq)
{
  const long idx = (long)blockIdx.x * 256 + threadIdx.x;
  const int p  = (int)(idx & (HW - 1));
  const long bc = idx >> 14;                 // b*576 + ch
  const int ch = (int)(bc % C3);
  const int y = p >> 7, x = p & 127;
  const _Float16* src = in + bc * HW;
  const float* wk = w + ch * 9;
  float s = 0.f;
#pragma unroll
  for (int dy = 0; dy < 3; ++dy) {
    const int yy = y + dy - 1;
    if (yy < 0 || yy > 127) continue;
#pragma unroll
    for (int dx = 0; dx < 3; ++dx) {
      const int xx = x + dx - 1;
      if (xx < 0 || xx > 127) continue;
      s += wk[dy * 3 + dx] * (float)src[yy * 128 + xx];
    }
  }
  out[idx] = (_Float16)s;

  // fused sum-of-squares for q,k channels (ch uniform across the block)
  if (ch < 2 * DIMC) {
    __shared__ float red[256];
    red[threadIdx.x] = s * s;
    __syncthreads();
    for (int off = 128; off > 0; off >>= 1) {
      if (threadIdx.x < off) red[threadIdx.x] += red[threadIdx.x + off];
      __syncthreads();
    }
    if (threadIdx.x == 0) {
      const int b = (int)(bc / C3);
      atomicAdd(&sumsq[b * 2 * DIMC + ch], red[0]);   // global_atomic_add_f32
    }
  }
}

// ---------------------------------------------------------------------------
// Per-(b,h): raw Gram S[32,32] = Q_h * K_h^T over K=16384 via WMMA, split
// across 8 waves, reduced in LDS (ds_add_f32). Then fused: reciprocal-norm
// scaling + temperature + softmax over the valid 24x24 block, and
// Weff[b][:, h*24+d] = W_proj[:, h*24+c] @ attn  (this head's columns only).
//
// Note: tile rows/cols 24..31 intentionally load neighboring (valid, finite)
// channels of the dw buffer without masking; they only pollute S entries that
// are never read. This keeps the hot loop free of exec-mask churn.
// ---------------------------------------------------------------------------
__global__ __launch_bounds__(256) void gram_softmax_weff_kernel(
    const _Float16* __restrict__ dwb, const float* __restrict__ sumsq,
    const float* __restrict__ temp, const float* __restrict__ wp,
    float* __restrict__ weff)
{
  const int bh = blockIdx.x, b = bh >> 3, h = bh & 7;
  const _Float16* Q  = dwb + ((long)b * C3 + h * HD) * HW;
  const _Float16* Kp = dwb + ((long)b * C3 + DIMC + h * HD) * HW;

  __shared__ float S[32 * 32];
  __shared__ float At[HD * HD];
  __shared__ float rk24[HD];
  const int tid = threadIdx.x, wave = tid >> 5, lane = tid & 31;
  const int ln = lane & 15, kb = (lane < 16) ? 0 : 8;
  for (int i = tid; i < 1024; i += 256) S[i] = 0.f;
  __syncthreads();

  v8f acc[2][2] = {};
  for (int kk = wave * 32; kk < HW; kk += 256) {
    v16h aq[2], bk[2];
#pragma unroll
    for (int i = 0; i < 2; ++i) {
      const int r = i * 16 + ln;              // q channel (A row)
      const v8h* pq = (const v8h*)(Q + (long)r * HW + kk + kb);
      v8h qlo = pq[0], qhi = pq[2];           // K: kb..kb+7, kb+16..kb+23
      const v8h* pk = (const v8h*)(Kp + (long)r * HW + kk + kb); // B column d=r
      v8h klo = pk[0], khi = pk[2];
      v16h f, g;
#pragma unroll
      for (int j = 0; j < 8; ++j) {
        f[j] = qlo[j]; f[8 + j] = qhi[j];
        g[j] = klo[j]; g[8 + j] = khi[j];
      }
      aq[i] = f;
      bk[i] = g;
    }
#pragma unroll
    for (int i = 0; i < 2; ++i)
#pragma unroll
      for (int j = 0; j < 2; ++j)
        acc[i][j] = __builtin_amdgcn_wmma_f32_16x16x32_f16(
            false, aq[i], false, bk[j], (short)0, acc[i][j], false, false);
  }

  // cross-wave reduction in LDS (ds_add_f32)
#pragma unroll
  for (int i = 0; i < 2; ++i)
#pragma unroll
    for (int j = 0; j < 2; ++j) {
      const int col = j * 16 + ln;
      const int rbase = i * 16 + ((lane < 16) ? 0 : 8);
#pragma unroll
      for (int r = 0; r < 8; ++r)
        atomicAdd(&S[(rbase + r) * 32 + col], acc[i][j][r]);
    }

  // reciprocal k-norms
  if (tid < HD)
    rk24[tid] = 1.f / fmaxf(sqrtf(sumsq[b * 2 * DIMC + DIMC + h * HD + tid]), 1e-12f);
  __syncthreads();

  // fused normalize + temperature + softmax (rows c<24, cols d<24)
  if (tid < HD) {
    const int c = tid;
    const float rq = 1.f / fmaxf(sqrtf(sumsq[b * 2 * DIMC + h * HD + c]), 1e-12f);
    const float tp = temp[h];
    float vals[HD], mx = -1e30f;
#pragma unroll
    for (int d = 0; d < HD; ++d) {
      float v = S[c * 32 + d] * rq * rk24[d] * tp;
      vals[d] = v;
      mx = fmaxf(mx, v);
    }
    float sum = 0.f;
#pragma unroll
    for (int d = 0; d < HD; ++d) { float e = __expf(vals[d] - mx); vals[d] = e; sum += e; }
    const float inv = 1.f / sum;
#pragma unroll
    for (int d = 0; d < HD; ++d) At[c * HD + d] = vals[d] * inv;
  }
  __syncthreads();

  // Weff for this head's 192x24 column block: Weff[o, h*24+d] = sum_c Wp[o,h*24+c]*At[c,d]
  for (int i = tid; i < DIMC * HD; i += 256) {
    const int o = i / HD, d = i % HD;
    const float* wr = wp + o * DIMC + h * HD;
    float s = 0.f;
#pragma unroll
    for (int c = 0; c < HD; ++c) s += wr[c] * At[c * HD + d];
    weff[(long)b * DIMC * DIMC + o * DIMC + h * HD + d] = s;
  }
}

// ---------------------------------------------------------------------------
// Launch
// ---------------------------------------------------------------------------
extern "C" void kernel_launch(void* const* d_in, const int* in_sizes, int n_in,
                              void* d_out, int out_size, void* d_ws, size_t ws_size,
                              hipStream_t stream) {
  const float* x      = (const float*)d_in[0];   // [8,192,128,128]
  const float* w_qkv  = (const float*)d_in[1];   // [576,192]
  const float* w_dw   = (const float*)d_in[2];   // [576,1,3,3]
  const float* temper = (const float*)d_in[3];   // [8]
  const float* w_proj = (const float*)d_in[4];   // [192,192]
  float* out = (float*)d_out;

  char* ws = (char*)d_ws;
  const size_t QKV_BYTES = (size_t)BATCH * C3 * HW * sizeof(_Float16);  // 151 MB
  _Float16* qkv   = (_Float16*)ws;
  _Float16* dwb   = (_Float16*)(ws + QKV_BYTES);
  float*    sumsq = (float*)(ws + 2 * QKV_BYTES);                 // 8*384 f32
  float*    weff  = (float*)(ws + 2 * QKV_BYTES + 16384);         // 8*192*192 f32

  // 1) qkv = w_qkv @ x            (WMMA, f32 B -> f16 out)
  gemm192_wmma<false, true><<<dim3(HW / 16, C3 / 64, BATCH), 32, 0, stream>>>(
      w_qkv, 0L, x, (long)DIMC * HW, qkv, (long)C3 * HW, HW);

  // 2) zero norm accumulators
  zero_kernel<<<(BATCH * 2 * DIMC + 255) / 256, 256, 0, stream>>>(
      sumsq, BATCH * 2 * DIMC);

  // 3) depthwise 3x3 + fused q/k sum-of-squares
  dwconv3x3_kernel<<<(BATCH * C3 * HW) / 256, 256, 0, stream>>>(
      qkv, w_dw, dwb, sumsq);

  // 4) Gram (WMMA over K=16384) + softmax + Weff = W_proj @ blockdiag(attn)
  gram_softmax_weff_kernel<<<BATCH * HEADS, 256, 0, stream>>>(
      dwb, sumsq, temper, w_proj, weff);

  // 5) out = Weff @ V             (WMMA, f16 B -> f32 out)
  gemm192_wmma<true, false><<<dim3(HW / 16, DIMC / 64, BATCH), 32, 0, stream>>>(
      weff, (long)DIMC * DIMC, dwb + (long)2 * DIMC * HW, (long)C3 * HW,
      out, (long)DIMC * HW, HW);
}
